// lstm_latent_68307159876123
// MI455X (gfx1250) — matrix operand order repaired and checked by reading, not compiled, not run on hardware
//
#include <hip/hip_runtime.h>
#include <hip/hip_bf16.h>

// ---------------------------------------------------------------------------
// Persistent pipelined 6-cell LSTM + 2 linears for MI455X (gfx1250, wave32).
// bf16 WMMA (v_wmma_f32_16x16x32_bf16), weights LDS-resident per WGP,
// h activations hand off between pipeline stages through a 4-deep global ring
// with agent-scope atomic flags.
// ---------------------------------------------------------------------------

typedef __attribute__((ext_vector_type(16))) __bf16   v16bf;
typedef __attribute__((ext_vector_type(8)))  float    v8f;
typedef __attribute__((ext_vector_type(4)))  unsigned u32x4;
typedef __attribute__((ext_vector_type(8)))  unsigned u32x8;

constexpr int B_     = 256;   // batch
constexpr int T_     = 256;   // sequence length
constexpr int RING   = 4;     // h ring depth (timesteps)
constexpr int SLICE  = 32;    // output features per workgroup

constexpr int DIN[6] = { 64, 128, 256, 512, 512, 512};
constexpr int DH [6] = {128, 256, 512, 512, 512, 256};
constexpr int KD [6] = {192, 384, 768,1024,1024, 768};
// workgroups per stage (cells 0..5, then linear stage)
constexpr int NWG[7]    = {4, 8, 16, 16, 16, 8, 1};
constexpr int WGBASE[7] = {0, 4, 12, 28, 44, 60, 68};
constexpr int TOTAL_WG  = 69;

// element offsets inside workspace sections
constexpr int WOFF[6] = {0, 98304, 491520, 2064384, 4161536, 6258688}; // bf16
constexpr int BOFF[6] = {0, 512, 1536, 3584, 5632, 7680};              // f32
constexpr int ROFF[6] = {0, 131072, 393216, 917504, 1441792, 1966080}; // bf16

// workspace byte offsets (all 256B aligned)
constexpr size_t WF_OFF    = 0;          // 7,045,120 bf16 = 14,090,240 B
constexpr size_t BIAS_OFF  = 14090240;   // 8704 f32
constexpr size_t XBF_OFF   = 14125056;   // 256*256*64 bf16
constexpr size_t RING_OFF  = 22513664;   // 2,228,224 bf16
constexpr size_t LINW_OFF  = 26970112;   // 40960 bf16 (w1 then w2)
constexpr size_t LINB_OFF  = 27052032;   // 192 f32 (b1 then b2)
constexpr size_t FLAGS_OFF = 27052800;   // 2*6*256 int (done, consumed)
constexpr size_t WS_BYTES  = 27065088;

constexpr int RING_U32     = 1114112;    // ring bytes / 4
constexpr unsigned MAX_DYN_LDS = 4*SLICE*1024*2 + B_*SLICE*4 + 4*SLICE*4; // 295,424

// ---------------------------------------------------------------------------

__device__ __forceinline__ v16bf ld2x8(const __bf16* p0, const __bf16* p1) {
  u32x4 lo = *(const u32x4*)p0;      // global_load_b128 / ds_load_b128
  u32x4 hi = *(const u32x4*)p1;
  u32x8 w;
  w[0]=lo[0]; w[1]=lo[1]; w[2]=lo[2]; w[3]=lo[3];
  w[4]=hi[0]; w[5]=hi[1]; w[6]=hi[2]; w[7]=hi[3];
  return __builtin_bit_cast(v16bf, w);
}

__device__ __forceinline__ v8f wmma_bf16(v16bf a, v16bf b, v8f c) {
  return __builtin_amdgcn_wmma_f32_16x16x32_bf16(
      /*neg_a=*/false, a, /*neg_b=*/false, b,
      /*c_mod=*/(short)0, c, /*reuse_a=*/false, /*reuse_b=*/false);
}

__device__ __forceinline__ float sigm(float x) {
  return 1.0f / (1.0f + __expf(-x));
}

__device__ __forceinline__ void spin_until(int* p, int target) {
  while (__hip_atomic_load(p, __ATOMIC_ACQUIRE, __HIP_MEMORY_SCOPE_AGENT) < target)
    __builtin_amdgcn_s_sleep(2);
}

// ------------------------------- prep kernels ------------------------------

__global__ void prep_weights(const float* __restrict__ Wih,
                             const float* __restrict__ Whh,
                             const float* __restrict__ bih,
                             const float* __restrict__ bhh,
                             __bf16* __restrict__ Wf,
                             float* __restrict__ bf,
                             int din, int dh) {
  const int K = din + dh;
  const int total = 4 * dh * K;
  int idx = blockIdx.x * 256 + threadIdx.x;
  if (idx < total) {
    int r = idx / K, k = idx - r * K;
    float v = (k < din) ? Wih[r * din + k] : Whh[r * dh + (k - din)];
    Wf[idx] = (__bf16)v;
  }
  if (idx < 4 * dh) bf[idx] = bih[idx] + bhh[idx];
}

__global__ void prep_x(const float* __restrict__ X, __bf16* __restrict__ Xbf) {
  int idx = blockIdx.x * 256 + threadIdx.x;
  if (idx < B_ * T_ * 64) Xbf[idx] = (__bf16)X[idx];
}

__global__ void prep_misc(unsigned* __restrict__ ringu,
                          int* __restrict__ flags,
                          const float* __restrict__ w1, const float* __restrict__ b1,
                          const float* __restrict__ w2, const float* __restrict__ b2,
                          __bf16* __restrict__ linw, float* __restrict__ linb) {
  const int gid = blockIdx.x * 256 + threadIdx.x;
  const int stride = gridDim.x * 256;
  for (int i = gid; i < RING_U32; i += stride) ringu[i] = 0u;
  for (int i = gid; i < 2 * 6 * T_; i += stride) flags[i] = 0;
  for (int i = gid; i < 40960; i += stride)
    linw[i] = (__bf16)((i < 32768) ? w1[i] : w2[i - 32768]);
  for (int i = gid; i < 192; i += stride)
    linb[i] = (i < 128) ? b1[i] : b2[i - 128];
}

// --------------------------- persistent pipeline ---------------------------

__global__ void __launch_bounds__(256, 1)
lstm_persistent(const __bf16* __restrict__ wf,
                const float*  __restrict__ biasf,
                const __bf16* __restrict__ xbf,
                __bf16* __restrict__ ring,
                const __bf16* __restrict__ linw,
                const float*  __restrict__ linb,
                int* __restrict__ flags,
                float* __restrict__ out) {
  extern __shared__ __align__(16) char smem[];
  const int tid  = threadIdx.x;
  const int wv   = tid >> 5;          // wave 0..7
  const int ln   = tid & 31;          // lane 0..31
  const int lnm  = ln & 15;           // lane % 16
  const int khi  = (ln >> 4);         // 0: k 0-7,16-23, 1: k 8-15,24-31 of each 32-step (A layout)
  int* doneF = flags;                 // [6][T]
  int* consF = flags + 6 * T_;        // [6][T]
  const int wg = blockIdx.x;

  if (wg < WGBASE[6]) {
    // ------------------------------ LSTM cell stage ------------------------
    int c = 0;
    while (wg >= WGBASE[c + 1]) ++c;
    const int s     = wg - WGBASE[c];
    const int din   = DIN[c], dh = DH[c], K = KD[c];
    const int dprev = (c > 0) ? DH[c - 1] : 64;
    const __bf16* Wc    = wf + WOFF[c];
    const float*  bc    = biasf + BOFF[c];
    __bf16*       ringC = ring + ROFF[c];
    const __bf16* ringP = (c > 0) ? (ring + ROFF[c - 1]) : nullptr;

    __bf16* w_lds = (__bf16*)smem;                                   // [128][K]
    float*  c_lds = (float*)(smem + (size_t)4 * SLICE * K * 2);      // [B][SLICE]
    float*  b_lds = c_lds + B_ * SLICE;                              // [128]

    { // stage weights (once): 128 rows of K bf16, 2 threads per row
      int row = tid >> 1;
      int g = row / SLICE, j = row % SLICE;
      const __bf16* src = Wc + (size_t)(g * dh + s * SLICE + j) * K;
      __bf16* dst = w_lds + (size_t)row * K;
      int k0 = (tid & 1) * (K / 2);
      for (int k = 0; k < K / 2; k += 8)
        *(u32x4*)(dst + k0 + k) = *(const u32x4*)(src + k0 + k);
    }
    if (tid < 128) {
      int g = tid / SLICE, j = tid % SLICE;
      b_lds[tid] = bc[g * dh + s * SLICE + j];
    }
    for (int i = tid; i < B_ * SLICE; i += 256) c_lds[i] = 0.0f;
    __syncthreads();

    const int m0 = wv * 32 + lnm;     // A-tile 0 row for this lane
    const int m1 = m0 + 16;           // A-tile 1 row

    for (int t = 0; t < T_; ++t) {
      if (tid == 0) {
        if (c > 0)     spin_until(&doneF[(c - 1) * T_ + t], NWG[c - 1]);
        if (t > 0)     spin_until(&doneF[c * T_ + (t - 1)], NWG[c]);
        if (t >= RING) spin_until(&consF[c * T_ + (t - RING)], NWG[c + 1]);
      }
      __syncthreads();

      const __bf16* xsrc = (c == 0) ? xbf
                         : (ringP + (size_t)(t % RING) * B_ * dprev);
      const __bf16* hsrc = ringC + (size_t)((t - 1 + RING) % RING) * B_ * dh;

      v8f acc[16];
#pragma unroll
      for (int i = 0; i < 16; ++i) acc[i] = (v8f)0.0f;

      for (int k0 = 0; k0 < K; k0 += 32) {
        const int kk = k0 + khi * 8;
        const __bf16 *pa0, *pa1;
        if (k0 < din) {
          if (c == 0) {
            pa0 = xsrc + ((size_t)m0 * T_ + t) * 64 + kk;
            pa1 = xsrc + ((size_t)m1 * T_ + t) * 64 + kk;
          } else {
            pa0 = xsrc + (size_t)m0 * dprev + kk;
            pa1 = xsrc + (size_t)m1 * dprev + kk;
          }
        } else {
          const int kh = kk - din;
          pa0 = hsrc + (size_t)m0 * dh + kh;
          pa1 = hsrc + (size_t)m1 * dh + kh;
        }
        v16bf a0 = ld2x8(pa0, pa0 + 16);
        v16bf a1 = ld2x8(pa1, pa1 + 16);
#pragma unroll
        for (int nt = 0; nt < 8; ++nt) {  // nt = gate*2 + fb
          const __bf16* pb = w_lds + (size_t)(nt * 16 + lnm) * K
                           + k0 + ((ln < 16) ? 0 : 8);
          v16bf b = ld2x8(pb, pb + 16);
          acc[nt]     = wmma_bf16(a0, b, acc[nt]);
          acc[8 + nt] = wmma_bf16(a1, b, acc[8 + nt]);
        }
      }

      // ------------- activations + state update + h handoff ---------------
      __bf16* hdst = ringC + (size_t)(t % RING) * B_ * dh;
#pragma unroll
      for (int mi = 0; mi < 2; ++mi) {
#pragma unroll
        for (int fb = 0; fb < 2; ++fb) {
          v8f gi = acc[mi * 8 + 0 * 2 + fb];
          v8f gf = acc[mi * 8 + 1 * 2 + fb];
          v8f gg = acc[mi * 8 + 2 * 2 + fb];
          v8f go = acc[mi * 8 + 3 * 2 + fb];
          const int feat = fb * 16 + lnm;
          const float bi = b_lds[0 * SLICE + feat];
          const float bf = b_lds[1 * SLICE + feat];
          const float bg = b_lds[2 * SLICE + feat];
          const float bo = b_lds[3 * SLICE + feat];
#pragma unroll
          for (int v = 0; v < 8; ++v) {
            const int m = (wv * 2 + mi) * 16 + v + ((ln < 16) ? 0 : 8);
            float iv = sigm(gi[v] + bi);
            float fv = sigm(gf[v] + bf);
            float gv = tanhf(gg[v] + bg);
            float ov = sigm(go[v] + bo);
            float cn = fv * c_lds[m * SLICE + feat] + iv * gv;
            c_lds[m * SLICE + feat] = cn;
            hdst[(size_t)m * dh + (s * SLICE + feat)] = (__bf16)(ov * tanhf(cn));
          }
        }
      }
      __threadfence();   // make h stores agent-visible
      __syncthreads();
      if (tid == 0) {
        __hip_atomic_fetch_add(&doneF[c * T_ + t], 1,
                               __ATOMIC_RELEASE, __HIP_MEMORY_SCOPE_AGENT);
        if (c > 0)
          __hip_atomic_fetch_add(&consF[(c - 1) * T_ + t], 1,
                                 __ATOMIC_RELEASE, __HIP_MEMORY_SCOPE_AGENT);
      }
    }
  } else {
    // ------------------------------ linear stage ---------------------------
    __bf16* w1_lds = (__bf16*)smem;              // [128][256]
    __bf16* w2_lds = w1_lds + 128 * 256;         // [64][128]
    __bf16* y1_lds = w2_lds + 64 * 128;          // [256][128]
    float*  b1_lds = (float*)(y1_lds + 256 * 128);
    float*  b2_lds = b1_lds + 128;

    for (int i = tid * 8; i < 128 * 256; i += 256 * 8)
      *(u32x4*)(w1_lds + i) = *(const u32x4*)(linw + i);
    for (int i = tid * 8; i < 64 * 128; i += 256 * 8)
      *(u32x4*)(w2_lds + i) = *(const u32x4*)(linw + 32768 + i);
    if (tid < 128) b1_lds[tid] = linb[tid];
    if (tid < 64)  b2_lds[tid] = linb[128 + tid];
    __syncthreads();

    const int m0 = wv * 32 + lnm;
    const int m1 = m0 + 16;

    for (int t = 0; t < T_; ++t) {
      if (tid == 0) spin_until(&doneF[5 * T_ + t], NWG[5]);
      __syncthreads();
      const __bf16* h5 = ring + ROFF[5] + (size_t)(t % RING) * B_ * 256;

      // GEMM1: [256,256] x [256,128]
      v8f acc[16];
#pragma unroll
      for (int i = 0; i < 16; ++i) acc[i] = (v8f)0.0f;
      for (int k0 = 0; k0 < 256; k0 += 32) {
        const int kk = k0 + khi * 8;
        v16bf a0 = ld2x8(h5 + (size_t)m0 * 256 + kk, h5 + (size_t)m0 * 256 + kk + 16);
        v16bf a1 = ld2x8(h5 + (size_t)m1 * 256 + kk, h5 + (size_t)m1 * 256 + kk + 16);
#pragma unroll
        for (int nt = 0; nt < 8; ++nt) {
          const __bf16* pb = w1_lds + (size_t)(nt * 16 + lnm) * 256
                           + k0 + ((ln < 16) ? 0 : 8);
          v16bf b = ld2x8(pb, pb + 16);
          acc[nt]     = wmma_bf16(a0, b, acc[nt]);
          acc[8 + nt] = wmma_bf16(a1, b, acc[8 + nt]);
        }
      }
#pragma unroll
      for (int mi = 0; mi < 2; ++mi)
#pragma unroll
        for (int nt = 0; nt < 8; ++nt) {
          const int n = nt * 16 + lnm;
          const float bb = b1_lds[n];
#pragma unroll
          for (int v = 0; v < 8; ++v) {
            const int m = (wv * 2 + mi) * 16 + v + ((ln < 16) ? 0 : 8);
            y1_lds[m * 128 + n] = (__bf16)(acc[mi * 8 + nt][v] + bb);
          }
        }
      __syncthreads();

      // GEMM2: [256,128] x [128,64]
      v8f acc2[8];
#pragma unroll
      for (int i = 0; i < 8; ++i) acc2[i] = (v8f)0.0f;
      for (int k0 = 0; k0 < 128; k0 += 32) {
        const int kk = k0 + khi * 8;
        v16bf a0 = ld2x8(y1_lds + (size_t)m0 * 128 + kk, y1_lds + (size_t)m0 * 128 + kk + 16);
        v16bf a1 = ld2x8(y1_lds + (size_t)m1 * 128 + kk, y1_lds + (size_t)m1 * 128 + kk + 16);
#pragma unroll
        for (int nt = 0; nt < 4; ++nt) {
          const __bf16* pb = w2_lds + (size_t)(nt * 16 + lnm) * 128
                           + k0 + ((ln < 16) ? 0 : 8);
          v16bf b = ld2x8(pb, pb + 16);
          acc2[nt]     = wmma_bf16(a0, b, acc2[nt]);
          acc2[4 + nt] = wmma_bf16(a1, b, acc2[4 + nt]);
        }
      }
#pragma unroll
      for (int mi = 0; mi < 2; ++mi)
#pragma unroll
        for (int nt = 0; nt < 4; ++nt) {
          const int n = nt * 16 + lnm;
          const float bb = b2_lds[n];
#pragma unroll
          for (int v = 0; v < 8; ++v) {
            const int m = (wv * 2 + mi) * 16 + v + ((ln < 16) ? 0 : 8);
            out[((size_t)t * B_ + m) * 64 + n] = acc2[mi * 4 + nt][v] + bb;
          }
        }
      __syncthreads();
      if (tid == 0)
        __hip_atomic_fetch_add(&consF[5 * T_ + t], 1,
                               __ATOMIC_RELEASE, __HIP_MEMORY_SCOPE_AGENT);
    }
  }
}

// ------------------------------- launcher ----------------------------------

extern "C" void kernel_launch(void* const* d_in, const int* in_sizes, int n_in,
                              void* d_out, int out_size, void* d_ws, size_t ws_size,
                              hipStream_t stream) {
  (void)in_sizes; (void)n_in; (void)out_size; (void)ws_size;
  char* ws = (char*)d_ws;
  __bf16* wf    = (__bf16*)(ws + WF_OFF);
  float*  biasf = (float*) (ws + BIAS_OFF);
  __bf16* xbfp  = (__bf16*)(ws + XBF_OFF);
  __bf16* ringp = (__bf16*)(ws + RING_OFF);
  __bf16* linw  = (__bf16*)(ws + LINW_OFF);
  float*  linb  = (float*) (ws + LINB_OFF);
  int*    flags = (int*)   (ws + FLAGS_OFF);

  const float* X = (const float*)d_in[0];
  for (int c = 0; c < 6; ++c) {
    const float* Wih = (const float*)d_in[1 + 4 * c];
    const float* Whh = (const float*)d_in[2 + 4 * c];
    const float* bih = (const float*)d_in[3 + 4 * c];
    const float* bhh = (const float*)d_in[4 + 4 * c];
    const int total = 4 * DH[c] * KD[c];
    prep_weights<<<(total + 255) / 256, 256, 0, stream>>>(
        Wih, Whh, bih, bhh, wf + WOFF[c], biasf + BOFF[c], DIN[c], DH[c]);
  }
  prep_x<<<(B_ * T_ * 64 + 255) / 256, 256, 0, stream>>>(X, xbfp);
  prep_misc<<<2048, 256, 0, stream>>>(
      (unsigned*)ringp, flags,
      (const float*)d_in[25], (const float*)d_in[26],
      (const float*)d_in[27], (const float*)d_in[28], linw, linb);

  (void)hipFuncSetAttribute(reinterpret_cast<const void*>(lstm_persistent),
                            hipFuncAttributeMaxDynamicSharedMemorySize,
                            (int)MAX_DYN_LDS);
  lstm_persistent<<<TOTAL_WG, 256, MAX_DYN_LDS, stream>>>(
      wf, biasf, xbfp, ringp, linw, linb, flags, (float*)d_out);
}
